// CrossAttention_51539607770
// MI455X (gfx1250) — matrix operand order repaired and checked by reading, not compiled, not run on hardware
//
#include <hip/hip_runtime.h>
#include <hip/hip_bf16.h>
#include <cstdint>

// ---------------- CDNA5 (gfx1250) WMMA types ----------------
typedef _Float16 half_t;
typedef __attribute__((ext_vector_type(8)))  _Float16 v8h;
typedef __attribute__((ext_vector_type(16))) _Float16 v16h;
typedef __attribute__((ext_vector_type(8)))  float    v8f;

#define DIM      768
#define HEADS    12
#define HEAD_DIM 64
#define NSEQ     1024
#define BATCH    2
#define QKV_LD   (3 * DIM)      // 2304
#define ROWS     (BATCH * NSEQ) // 2048

union Frag { v16h v; v8h h[2]; };

static __device__ __forceinline__ v8f wmma_f16(v16h a, v16h b, v8f c) {
  // D = A(16x32 f16) * B(32x16 f16) + C(16x16 f32)
  return __builtin_amdgcn_wmma_f32_16x16x32_f16(false, a, false, b, (short)0, c,
                                                false, false);
}

static __device__ __forceinline__ v8f zero8() {
  v8f z;
  for (int i = 0; i < 8; ++i) z[i] = 0.f;
  return z;
}

// CDNA5 async global->LDS copy (ASYNCcnt-tracked, CDNA5 ISA).
// vdst = LDS byte address (low 32 bits of generic shared pointer), 64-bit vaddr.
static __device__ __forceinline__ void async_ld16(uint32_t lds_addr, const void* gptr) {
  asm volatile("global_load_async_to_lds_b128 %0, %1, off"
               :: "v"(lds_addr), "v"((uint64_t)(uintptr_t)gptr)
               : "memory");
}

// ---------------- f32 -> f16 conversion ----------------
__global__ void k_cvt(const float* __restrict__ in, half_t* __restrict__ out, int n) {
  int i = blockIdx.x * blockDim.x + threadIdx.x;
  if (i < n) out[i] = (half_t)in[i];
}

// W [K x N] f32 row-major  ->  Wt [N x K] f16  (so B fragments load contiguously)
__global__ void k_cvt_T(const float* __restrict__ in, half_t* __restrict__ out,
                        int K, int N) {
  int i = blockIdx.x * blockDim.x + threadIdx.x;
  if (i < K * N) {
    int k = i / N, n = i % N;
    out[(size_t)n * K + k] = (half_t)in[i];
  }
}

// ---------------- block-cooperative WMMA GEMM ----------------
// C[MxN] = A[MxK] * Bt[NxK]^T. Block = 4 waves = 64x64 C tile; wave w owns rows
// [m0+16w, +16). The shared 64x32 B tile is double-buffered in LDS and filled
// with GLOBAL_LOAD_ASYNC_TO_LDS_B128; next tile's fill overlaps current WMMAs.
// Grid is sized exactly (no partial blocks) so EXEC is all-1s at every WMMA and
// every thread reaches the barriers.
__global__ __launch_bounds__(128) void k_gemm(
    const half_t* __restrict__ A, const half_t* __restrict__ BT,
    half_t* __restrict__ Ch, float* __restrict__ Cf, const float* __restrict__ bias,
    half_t* __restrict__ vT, int M, int N, int K, int lda, int ldc)
{
  __shared__ __align__(16) half_t Bt[2][64][40];   // 80B row stride: banks spread
  const int t = threadIdx.x;
  const int lane = t & 31, wIn = t >> 5;
  const int ntb = N >> 6;
  const int m0 = ((blockIdx.x / ntb) << 6) + (wIn << 4);
  const int n0 = (blockIdx.x % ntb) << 6;
  const int g = lane >> 4, mr = lane & 15;
  const uint32_t ldsBase = (uint32_t)(uintptr_t)(&Bt[0][0][0]);
  const uint32_t bufStride = 64u * 40u * 2u;       // 5120 B per buffer

  // tile fill: 64 rows x 64 B = 256 16B-chunks = 2 per thread
  const int rowc = t >> 2, colc = t & 3;

  // preload buffer 0 (k = 0)
  {
    const half_t* g0 = BT + (size_t)(n0 + rowc) * K + colc * 8;
    async_ld16(ldsBase + rowc * 80 + colc * 16, g0);
    const half_t* g1 = BT + (size_t)(n0 + rowc + 32) * K + colc * 8;
    async_ld16(ldsBase + (rowc + 32) * 80 + colc * 16, g1);
  }

  v8f acc[4];
#pragma unroll
  for (int c = 0; c < 4; ++c) acc[c] = zero8();

  const half_t* arow = A + (size_t)(m0 + mr) * lda;

  for (int k = 0; k < K; k += 32) {
    const int cur = (k >> 5) & 1;
    const uint32_t curBase = ldsBase + (uint32_t)cur * bufStride; (void)curBase;
    if (k + 32 < K) {
      const uint32_t nb = ldsBase + (uint32_t)(cur ^ 1) * bufStride;
      const half_t* g0 = BT + (size_t)(n0 + rowc) * K + (k + 32) + colc * 8;
      async_ld16(nb + rowc * 80 + colc * 16, g0);
      const half_t* g1 = BT + (size_t)(n0 + rowc + 32) * K + (k + 32) + colc * 8;
      async_ld16(nb + (rowc + 32) * 80 + colc * 16, g1);
      asm volatile("s_wait_asynccnt 0x2" ::: "memory");  // current buffer landed
    } else {
      asm volatile("s_wait_asynccnt 0x0" ::: "memory");
    }
    __syncthreads();                                     // all waves' fills done

    Frag a;                                  // A 16x32: M=mr, K=16g+j
    a.h[0] = *(const v8h*)(arow + k + 16 * g);
    a.h[1] = *(const v8h*)(arow + k + 16 * g + 8);
    Frag bfr[4];                             // B 32x16 from LDS: N=mr, K=16g+j
#pragma unroll
    for (int c = 0; c < 4; ++c) {
      const half_t* bp = &Bt[cur][c * 16 + mr][16 * g];
      bfr[c].h[0] = *(const v8h*)(bp);
      bfr[c].h[1] = *(const v8h*)(bp + 8);
    }
#pragma unroll
    for (int c = 0; c < 4; ++c) acc[c] = wmma_f16(a.v, bfr[c].v, acc[c]);

    __syncthreads();               // done reading cur; safe to overwrite next iter
  }

#pragma unroll
  for (int c = 0; c < 4; ++c) {
#pragma unroll
    for (int r = 0; r < 8; ++r) {            // C: M=r+8g, N=mr
      int row = m0 + r + 8 * g;
      int col = n0 + c * 16 + mr;
      float val = acc[c][r];
      if (Cf) Cf[(size_t)row * ldc + col] = val + (bias ? bias[col] : 0.f);
      if (Ch) Ch[(size_t)row * ldc + col] = (half_t)val;
      if (vT && col >= 2 * DIM) {            // fused V^T scatter: vT[b][h][d][n]
        int hv = (col - 2 * DIM) >> 6;
        int dv = (col - 2 * DIM) & 63;
        int bb = row >> 10, nn = row & 1023;
        vT[(size_t)(((bb * HEADS + hv) << 6) + dv) * NSEQ + nn] = (half_t)val;
      }
    }
  }
}

// ---------------- fused dual-pass flash attention ----------------
// wave = (s, b, h, qtile): o_s[16 q-rows] = attn(q_s,k_s,v_s) + attn(q_s,k_!s,v_!s)
__global__ __launch_bounds__(128) void k_attn(
    const half_t* __restrict__ qkvA, const half_t* __restrict__ qkvB,
    const half_t* __restrict__ vtA,  const half_t* __restrict__ vtB,
    half_t* __restrict__ oA, half_t* __restrict__ oB)
{
  __shared__ __align__(16) _Float16 Pst[4][16][40];  // P staging, padded rows (80B)
  const int lane = threadIdx.x & 31;
  const int wIn  = threadIdx.x >> 5;
  const int wave = blockIdx.x * 4 + wIn;
  if (wave >= 2 * BATCH * HEADS * (NSEQ / 16)) return;
  const int s  = wave / (BATCH * HEADS * 64);
  int rem = wave % (BATCH * HEADS * 64);
  const int b  = rem / (HEADS * 64); rem %= HEADS * 64;
  const int h  = rem / 64;
  const int qt = rem % 64;
  const int g = lane >> 4, mr = lane & 15;

  const half_t* qkvS[2] = {qkvA, qkvB};
  const half_t* vtS[2]  = {vtA, vtB};
  half_t* oS[2] = {oA, oB};

  // resident Q fragments: 16x64 = two 16x32 A fragments
  const half_t* qrow = qkvS[s] + (size_t)(b * NSEQ + qt * 16 + mr) * QKV_LD + h * 64;
  Frag aq[2];
#pragma unroll
  for (int dk = 0; dk < 2; ++dk) {
    aq[dk].h[0] = *(const v8h*)(qrow + dk * 32 + 16 * g);
    aq[dk].h[1] = *(const v8h*)(qrow + dk * 32 + 16 * g + 8);
  }

  v8f ofin[4];
#pragma unroll
  for (int c = 0; c < 4; ++c) ofin[c] = zero8();

  for (int p = 0; p < 2; ++p) {
    const int kv = p ? (1 - s) : s;
    const half_t* kb = qkvS[kv] + (size_t)(b * NSEQ) * QKV_LD + DIM + h * 64;
    const half_t* vb = vtS[kv] + (size_t)((b * HEADS + h) * 64) * NSEQ;

    float mrow[8], lrow[8];
    v8f oacc[4];
#pragma unroll
    for (int r = 0; r < 8; ++r) { mrow[r] = -3.0e38f; lrow[r] = 0.f; }
#pragma unroll
    for (int c = 0; c < 4; ++c) oacc[c] = zero8();

    for (int j = 0; j < NSEQ / 32; ++j) {
      // hoist V fragments: independent loads overlap S WMMAs + softmax below
      Frag bv[4];
#pragma unroll
      for (int c = 0; c < 4; ++c) {
        const half_t* vp = vb + (size_t)(c * 16 + mr) * NSEQ + j * 32 + 16 * g;
        bv[c].h[0] = *(const v8h*)(vp);
        bv[c].h[1] = *(const v8h*)(vp + 8);
      }
      // S[16x32] = Q(16x64) * K^T(64x32) : B frag lane -> one k-row, contiguous d
      v8f sacc[2];
#pragma unroll
      for (int kt = 0; kt < 2; ++kt) {
        const half_t* kp = kb + (size_t)(j * 32 + kt * 16 + mr) * QKV_LD;
        Frag b0, b1;
        b0.h[0] = *(const v8h*)(kp + 16 * g);
        b0.h[1] = *(const v8h*)(kp + 16 * g + 8);
        b1.h[0] = *(const v8h*)(kp + 32 + 16 * g);
        b1.h[1] = *(const v8h*)(kp + 32 + 16 * g + 8);
        v8f tz = wmma_f16(aq[0].v, b0.v, zero8());
        sacc[kt] = wmma_f16(aq[1].v, b1.v, tz);
      }
      // online softmax: each row lives in one VGPR across a 16-lane half-wave
#pragma unroll
      for (int r = 0; r < 8; ++r) {
        float s0 = sacc[0][r] * 0.125f;   // SCALE = HEAD_DIM^-0.5
        float s1 = sacc[1][r] * 0.125f;
        float tm = fmaxf(s0, s1);
        tm = fmaxf(tm, __shfl_xor(tm, 1));
        tm = fmaxf(tm, __shfl_xor(tm, 2));
        tm = fmaxf(tm, __shfl_xor(tm, 4));
        tm = fmaxf(tm, __shfl_xor(tm, 8));
        float mnew = fmaxf(mrow[r], tm);
        float sc = __expf(mrow[r] - mnew);
        mrow[r] = mnew;
        lrow[r] *= sc;
#pragma unroll
        for (int c = 0; c < 4; ++c) oacc[c][r] *= sc;
        float p0 = __expf(s0 - mnew);
        float p1 = __expf(s1 - mnew);
        lrow[r] += p0 + p1;               // per-lane partial, reduced once at end
        Pst[wIn][r + 8 * g][mr]      = (half_t)p0;  // C-layout -> LDS
        Pst[wIn][r + 8 * g][16 + mr] = (half_t)p1;
      }
      asm volatile("s_wait_dscnt 0x0" ::: "memory");
      Frag ap;                            // reload P as A fragment (16x32)
      const _Float16* pr = &Pst[wIn][mr][0];
      ap.h[0] = *(const v8h*)(pr + 16 * g);
      ap.h[1] = *(const v8h*)(pr + 16 * g + 8);
      // O += P(16x32) * V(32x64) using the prefetched V fragments
#pragma unroll
      for (int c = 0; c < 4; ++c) oacc[c] = wmma_f16(ap.v, bv[c].v, oacc[c]);
    }
    // finalize pass: full row-sum, normalize, accumulate into summed output
#pragma unroll
    for (int r = 0; r < 8; ++r) {
      float l = lrow[r];
      l += __shfl_xor(l, 1);
      l += __shfl_xor(l, 2);
      l += __shfl_xor(l, 4);
      l += __shfl_xor(l, 8);
      float inv = 1.0f / l;
#pragma unroll
      for (int c = 0; c < 4; ++c) ofin[c][r] += oacc[c][r] * inv;
    }
  }
  // store o[b][n][h*64 + d] as f16 for the projection GEMM
  half_t* ob = oS[s];
#pragma unroll
  for (int c = 0; c < 4; ++c)
#pragma unroll
    for (int r = 0; r < 8; ++r) {
      int n = qt * 16 + r + 8 * g;
      ob[(size_t)(b * NSEQ + n) * DIM + h * 64 + c * 16 + mr] = (half_t)ofin[c][r];
    }
}

// ---------------- host-side orchestration ----------------
extern "C" void kernel_launch(void* const* d_in, const int* in_sizes, int n_in,
                              void* d_out, int out_size, void* d_ws, size_t ws_size,
                              hipStream_t stream) {
  (void)in_sizes; (void)n_in; (void)out_size;
  const float* x[2]    = {(const float*)d_in[0], (const float*)d_in[1]};
  const float* Wqkv[2] = {(const float*)d_in[2], (const float*)d_in[3]};
  const float* Wp[2]   = {(const float*)d_in[4], (const float*)d_in[6]};
  const float* bp[2]   = {(const float*)d_in[5], (const float*)d_in[7]};

  char* ws = (char*)d_ws;
  size_t off = 0;
  auto alloc = [&](size_t bytes) -> half_t* {
    half_t* p = (half_t*)(ws + off);
    off = (off + bytes + 255) & ~(size_t)255;
    return p;
  };
  half_t *xh[2], *wqkvT[2], *wpT[2], *qkvh[2], *vTh[2], *oh[2];
  for (int s = 0; s < 2; ++s) xh[s]    = alloc((size_t)ROWS * DIM * 2);
  for (int s = 0; s < 2; ++s) wqkvT[s] = alloc((size_t)QKV_LD * DIM * 2);
  for (int s = 0; s < 2; ++s) wpT[s]   = alloc((size_t)DIM * DIM * 2);
  for (int s = 0; s < 2; ++s) qkvh[s]  = alloc((size_t)ROWS * QKV_LD * 2);
  for (int s = 0; s < 2; ++s) vTh[s]   = alloc((size_t)BATCH * HEADS * HEAD_DIM * NSEQ * 2);
  for (int s = 0; s < 2; ++s) oh[s]    = alloc((size_t)ROWS * DIM * 2);
  if (off > ws_size) return;  // avoid OOB if workspace too small (deterministic)

  // 1) precision conversion (+ weight transposes)
  for (int s = 0; s < 2; ++s) {
    int nx = ROWS * DIM;
    k_cvt<<<(nx + 255) / 256, 256, 0, stream>>>(x[s], xh[s], nx);
    int nw = DIM * QKV_LD;
    k_cvt_T<<<(nw + 255) / 256, 256, 0, stream>>>(Wqkv[s], wqkvT[s], DIM, QKV_LD);
    int np = DIM * DIM;
    k_cvt_T<<<(np + 255) / 256, 256, 0, stream>>>(Wp[s], wpT[s], DIM, DIM);
  }
  // 2) QKV projection GEMMs (f16 out + fused V^T scatter); 64x64 tile per block
  for (int s = 0; s < 2; ++s) {
    int blocks = (ROWS / 64) * (QKV_LD / 64);          // 32 * 36 = 1152 (exact)
    k_gemm<<<blocks, 128, 0, stream>>>(
        xh[s], wqkvT[s], qkvh[s], nullptr, nullptr, vTh[s],
        ROWS, QKV_LD, DIM, DIM, QKV_LD);
  }
  // 3) fused four-way attention (3072 waves)
  k_attn<<<(2 * BATCH * HEADS * (NSEQ / 16)) / 4, 128, 0, stream>>>(
      qkvh[0], qkvh[1], vTh[0], vTh[1], oh[0], oh[1]);
  // 4) output projections with bias -> f32 d_out (y1 then y2)
  for (int s = 0; s < 2; ++s) {
    int blocks = (ROWS / 64) * (DIM / 64);             // 32 * 12 = 384 (exact)
    k_gemm<<<blocks, 128, 0, stream>>>(
        oh[s], wpT[s], nullptr, (float*)d_out + (size_t)s * ROWS * DIM, bp[s],
        nullptr, ROWS, DIM, DIM, DIM, DIM);
  }
}
